// Net_90151363543457
// MI455X (gfx1250) — matrix operand order repaired and checked by reading, not compiled, not run on hardware
//
#include <hip/hip_runtime.h>
#include <hip/hip_bf16.h>
#include <math.h>

#define N_NODES 50000
#define E_EDGES 800000
#define F_DIM   768
#define H_DIM   256
#define G_GR    512
#define C_CLS   2
#define NHID    128

typedef __attribute__((ext_vector_type(16))) __bf16    v16bf;
typedef __attribute__((ext_vector_type(8)))  float     v8f;
typedef __attribute__((ext_vector_type(8)))  unsigned  u32x8;

#if defined(__has_builtin)
#if __has_builtin(__builtin_amdgcn_global_load_async_to_lds_b128) && \
    __has_builtin(__builtin_amdgcn_s_wait_asynccnt)
#define HAVE_ASYNC_LDS 1
#endif
#endif

__device__ __forceinline__ unsigned short f2bf(float f) {
  unsigned u = __float_as_uint(f);
  u += 0x7FFFu + ((u >> 16) & 1u);          // round-to-nearest-even
  return (unsigned short)(u >> 16);
}
__device__ __forceinline__ unsigned pack2bf(float a, float b) {
  return (unsigned)f2bf(a) | ((unsigned)f2bf(b) << 16);
}
__device__ __forceinline__ float selu_f(float x) {
  const float scale = 1.0507009873554805f, alpha = 1.6732632423543772f;
  return scale * (x > 0.f ? x : alpha * (expf(x) - 1.f));
}
__device__ __forceinline__ void atomicMaxF(float* addr, float v) {
  if (v >= 0.f) atomicMax((int*)addr, __float_as_int(v));
  else          atomicMin((unsigned int*)addr, __float_as_uint(v));
}

#if HAVE_ASYNC_LDS
typedef __attribute__((ext_vector_type(4))) int i32x4;
typedef __attribute__((address_space(1))) i32x4* gas_v4i;
typedef __attribute__((address_space(3))) i32x4* las_v4i;
__device__ __forceinline__ void async_copy16(const unsigned* g, unsigned* l) {
  // generic->AS1: same 64-bit value; generic->AS3: low 32 bits are LDS offset
  __builtin_amdgcn_global_load_async_to_lds_b128(
      (gas_v4i)(uintptr_t)(g),
      (las_v4i)(unsigned)(uintptr_t)(l),
      0, 0);
}
#endif

// ---------------------------------------------------------------------------
// Packed-bf16 WMMA GEMM: C[M,256] = A x B, f32 accumulate.
//   A16: [M][K2]  row-major bf16 pairs  (K2 = K/2 uints per row)
//   Bt : [256][K2] per-column bf16 K-pairs (pre-transposed weights)
// 128x128 block tile, 16 kpairs (K=32) per step, 8 waves (4x2), wave = 32x64.
// LDS stride 20 uints: 16B-aligned staging stores, conflict-free frag loads.
// Output width fixed at 256: store addresses fold to immediate offsets.
// ---------------------------------------------------------------------------
#define LDS_STRIDE 20
#define NCOL 256

__global__ void __launch_bounds__(256) gemm_bf16_wmma(
    const unsigned* __restrict__ A16, const unsigned* __restrict__ Bt,
    float* __restrict__ C, int M, int K2) {
  __shared__ unsigned aT[128 * LDS_STRIDE];   // [row][kpair]
  __shared__ unsigned bT[128 * LDS_STRIDE];   // [col][kpair]
  const int tid  = threadIdx.x;
  const int lane = tid & 31;
  const int wave = tid >> 5;
  const int wm   = wave & 3;          // 32-row strip
  const int wn   = wave >> 2;         // 64-col strip
  const int mBase = blockIdx.x * 128;
  const int nBase = blockIdx.y * 128;
  const int half = lane >> 4;
  const int l15  = lane & 15;

  v8f acc[2][4] = {};

  for (int kp0 = 0; kp0 < K2; kp0 += 16) {
    __syncthreads();
    // ---- stage A tile: 128 rows x 16 kpairs, 16B chunks
#pragma unroll
    for (int j = 0; j < 2; ++j) {
      int idx = j * 256 + tid;            // 512 chunks
      int row = idx >> 2;                 // 0..127
      int c   = idx & 3;
      int gr  = mBase + row;
      int grc = (gr < M) ? gr : (M - 1);  // clamp: rows >= M never stored
      const unsigned* gp = A16 + (size_t)grc * K2 + kp0 + 4 * c;
      unsigned* lp = &aT[row * LDS_STRIDE + 4 * c];
#if HAVE_ASYNC_LDS
      async_copy16(gp, lp);
#else
      *(uint4*)lp = *(const uint4*)gp;
#endif
    }
    // ---- stage B tile: 128 cols x 16 kpairs, 16B chunks
#pragma unroll
    for (int j = 0; j < 2; ++j) {
      int idx = j * 256 + tid;
      int col = idx >> 2;
      int c   = idx & 3;
      const unsigned* gp = Bt + (size_t)(nBase + col) * K2 + kp0 + 4 * c;
      unsigned* lp = &bT[col * LDS_STRIDE + 4 * c];
#if HAVE_ASYNC_LDS
      async_copy16(gp, lp);
#else
      *(uint4*)lp = *(const uint4*)gp;
#endif
    }
#if HAVE_ASYNC_LDS
    __builtin_amdgcn_s_wait_asynccnt(0);
#endif
    __syncthreads();

    // A frag: 16x32 bf16. lane row = l15; kpair = (p&3) + 4*half + 8*(p>>2)
    v16bf afrag[2];
#pragma unroll
    for (int mi = 0; mi < 2; ++mi) {
      const unsigned* rp = aT + (wm * 32 + mi * 16 + l15) * LDS_STRIDE;
      u32x8 u;
#pragma unroll
      for (int p = 0; p < 8; ++p)
        u[p] = rp[(p & 3) + 4 * half + 8 * (p >> 2)];
      afrag[mi] = __builtin_bit_cast(v16bf, u);
    }
    // B frag: 32x16 bf16. lane col = l15; kpair = (p&3) + 4*half + 8*(p>>2)
    v16bf bfrag[4];
#pragma unroll
    for (int ni = 0; ni < 4; ++ni) {
      const unsigned* cp = bT + (wn * 64 + ni * 16 + l15) * LDS_STRIDE;
      u32x8 u;
#pragma unroll
      for (int p = 0; p < 8; ++p)
        u[p] = cp[(p & 3) + 4 * half + 8 * (p >> 2)];
      bfrag[ni] = __builtin_bit_cast(v16bf, u);
    }
#pragma unroll
    for (int mi = 0; mi < 2; ++mi)
#pragma unroll
      for (int ni = 0; ni < 4; ++ni)
        acc[mi][ni] = __builtin_amdgcn_wmma_f32_16x16x32_bf16(
            false, afrag[mi], false, bfrag[ni], (short)0, acc[mi][ni],
            false, false);
  }

  // D layout: vgpr r -> row = r + 8*half; col = l15
  if (mBase + 128 <= M) {
    // full block: no per-element guards; row stride folds to imm offsets
#pragma unroll
    for (int mi = 0; mi < 2; ++mi) {
#pragma unroll
      for (int ni = 0; ni < 4; ++ni) {
        int row0 = mBase + wm * 32 + mi * 16 + 8 * half;
        int col  = nBase + wn * 64 + ni * 16 + l15;
        float* cp = C + ((size_t)row0 << 8) + col;
#pragma unroll
        for (int r = 0; r < 8; ++r)
          cp[r << 8] = acc[mi][ni][r];
      }
    }
  } else {
    // tail block only
#pragma unroll
    for (int mi = 0; mi < 2; ++mi) {
#pragma unroll
      for (int ni = 0; ni < 4; ++ni) {
        int row0 = mBase + wm * 32 + mi * 16 + 8 * half;
        int col  = nBase + wn * 64 + ni * 16 + l15;
#pragma unroll
        for (int r = 0; r < 8; ++r) {
          int row = row0 + r;
          if (row < M) C[((size_t)row << 8) + col] = acc[mi][ni][r];
        }
      }
    }
  }
}

// ---------------------------------------------------------------------------
// conversion pre-passes
// ---------------------------------------------------------------------------
// fp32 row-major -> packed bf16 pairs (same layout, half the bytes)
__global__ void cvt_pairs(const float* __restrict__ in, unsigned* __restrict__ out,
                          int npairs) {
  int i = blockIdx.x * blockDim.x + threadIdx.x;
  if (i >= npairs) return;
  float2 v = *(const float2*)(in + 2 * i);
  out[i] = pack2bf(v.x, v.y);
}

// W[K][256] fp32 -> Bt[256][K2] packed per-column K-pairs
__global__ void cvt_bt(const float* __restrict__ W, unsigned* __restrict__ Bt,
                       int K2, int Nc) {
  int i = blockIdx.x * blockDim.x + threadIdx.x;
  if (i >= Nc * K2) return;
  int n = i / K2, kp = i - n * K2;
  Bt[i] = pack2bf(W[(2 * kp) * Nc + n], W[(2 * kp + 1) * Nc + n]);
}

// ---------------------------------------------------------------------------
// helpers / graph kernels
// ---------------------------------------------------------------------------
__global__ void initF(float* __restrict__ p, int n, float v) {
  int i = blockIdx.x * blockDim.x + threadIdx.x;
  if (i < n) p[i] = v;
}

__global__ void dots_k(const float* __restrict__ h, const float* __restrict__ a_s,
                       const float* __restrict__ a_d, float* __restrict__ outS,
                       float* __restrict__ outD) {
  int gid = blockIdx.x * blockDim.x + threadIdx.x;
  int node = gid >> 5, lane = gid & 31;
  if (node >= N_NODES) return;
  const float* row = h + (size_t)node * H_DIM;
  float ss = 0.f, sd = 0.f;
  for (int c = lane; c < H_DIM; c += 32) {
    float v = row[c];
    ss += v * a_s[c];
    sd += v * a_d[c];
  }
  for (int off = 16; off; off >>= 1) {
    ss += __shfl_xor(ss, off, 32);
    sd += __shfl_xor(sd, off, 32);
  }
  if (lane == 0) { outS[node] = ss; outD[node] = sd; }
}

__global__ void edge_e_max(const int* __restrict__ ei, const float* __restrict__ asd,
                           const float* __restrict__ add, float* __restrict__ e,
                           float* __restrict__ m) {
  int i = blockIdx.x * blockDim.x + threadIdx.x;
  if (i >= E_EDGES + N_NODES) return;
  int s = (i < E_EDGES) ? ei[i]           : (i - E_EDGES);
  int d = (i < E_EDGES) ? ei[E_EDGES + i] : (i - E_EDGES);
  float v = asd[s] + add[d];
  v = (v > 0.f) ? v : 0.2f * v;           // leaky_relu(0.2)
  e[i] = v;
  atomicMaxF(m + d, v);
}

__global__ void edge_exp_sum(const int* __restrict__ ei, float* __restrict__ e,
                             const float* __restrict__ m, float* __restrict__ denom) {
  int i = blockIdx.x * blockDim.x + threadIdx.x;
  if (i >= E_EDGES + N_NODES) return;
  int d = (i < E_EDGES) ? ei[E_EDGES + i] : (i - E_EDGES);
  float v = expf(e[i] - m[d]);
  e[i] = v;                               // in-place: e -> ex
  atomicAdd(&denom[d], v);
}

// one wave32 per edge; lanes cover the 256 channels; atomics land in L2
__global__ void edge_agg(const int* __restrict__ ei, const float* __restrict__ ex,
                         const float* __restrict__ h, float* __restrict__ agg) {
  int gid = blockIdx.x * blockDim.x + threadIdx.x;
  int edge = gid >> 5, lane = gid & 31;
  if (edge >= E_EDGES + N_NODES) return;
  int s = (edge < E_EDGES) ? ei[edge]           : (edge - E_EDGES);
  int d = (edge < E_EDGES) ? ei[E_EDGES + edge] : (edge - E_EDGES);
  float w = ex[edge];
  const float* hs = h + (size_t)s * H_DIM;
  float* ad = agg + (size_t)d * H_DIM;
#pragma unroll
  for (int j = 0; j < H_DIM / 32; ++j) {
    int c = lane + 32 * j;
    atomicAdd(ad + c, w * hs[c]);
  }
}

__global__ void norm_selu(float* __restrict__ agg, const float* __restrict__ denom,
                          const float* __restrict__ bias) {
  int i = blockIdx.x * blockDim.x + threadIdx.x;
  if (i >= N_NODES * H_DIM) return;
  int node = i >> 8, c = i & 255;
  agg[i] = selu_f(agg[i] / denom[node] + bias[c]);
}

__global__ void pool_accum(const float* __restrict__ h, const int* __restrict__ batch,
                           float* __restrict__ pooled, float* __restrict__ cnt) {
  int i = blockIdx.x * blockDim.x + threadIdx.x;
  if (i >= N_NODES * H_DIM) return;
  int node = i >> 8, c = i & 255;
  int g = batch[node];
  atomicAdd(&pooled[g * H_DIM + c], h[i]);
  if (c == 0) atomicAdd(&cnt[g], 1.f);
}

__global__ void pool_finish(float* __restrict__ pooled, const float* __restrict__ cnt) {
  int i = blockIdx.x * blockDim.x + threadIdx.x;
  if (i >= G_GR * H_DIM) return;
  pooled[i] = selu_f(pooled[i] / cnt[i >> 8]);
}

__global__ void first_idx_k(const int* __restrict__ batch, int* __restrict__ first) {
  int i = blockIdx.x * blockDim.x + threadIdx.x;
  if (i >= N_NODES) return;
  int g = batch[i];
  if (i == 0 || batch[i - 1] != g) first[g] = i;   // batch is sorted
}

__global__ void fc1_k(const float* __restrict__ pooled, const float* __restrict__ Wf1,
                      const float* __restrict__ bf1, float* __restrict__ z1) {
  int i = blockIdx.x * blockDim.x + threadIdx.x;
  if (i >= G_GR * NHID) return;
  int g = i >> 7, o = i & 127;
  const float* pr = pooled + g * H_DIM;
  float s = bf1[o];
  for (int k = 0; k < H_DIM; ++k) s += pr[k] * Wf1[k * NHID + o];
  z1[i] = selu_f(s);
}

__global__ void news_k(const float* __restrict__ x, const int* __restrict__ first,
                       const float* __restrict__ W0, const float* __restrict__ b0,
                       float* __restrict__ news) {
  int i = blockIdx.x * blockDim.x + threadIdx.x;
  if (i >= G_GR * NHID) return;
  int g = i >> 7, o = i & 127;
  const float* xr = x + (size_t)first[g] * F_DIM;
  float s = b0[o];
  for (int k = 0; k < F_DIM; ++k) s += xr[k] * W0[k * NHID + o];
  news[i] = fmaxf(s, 0.f);
}

__global__ void fccat_k(const float* __restrict__ z1, const float* __restrict__ news,
                        const float* __restrict__ Wf1, const float* __restrict__ bf1,
                        float* __restrict__ z2) {
  int i = blockIdx.x * blockDim.x + threadIdx.x;
  if (i >= G_GR * NHID) return;
  int g = i >> 7, o = i & 127;
  float s = bf1[o];
  for (int k = 0; k < NHID; ++k) s += z1[g * NHID + k] * Wf1[k * NHID + o];
  for (int k = 0; k < NHID; ++k) s += news[g * NHID + k] * Wf1[(NHID + k) * NHID + o];
  z2[i] = fmaxf(s, 0.f);
}

__global__ void out_k(const float* __restrict__ z2, const float* __restrict__ Wf2,
                      const float* __restrict__ bf2, float* __restrict__ out) {
  int g = blockIdx.x * blockDim.x + threadIdx.x;
  if (g >= G_GR) return;
  float l0 = bf2[0], l1 = bf2[1];
  for (int k = 0; k < NHID; ++k) {
    float z = z2[g * NHID + k];
    l0 += z * Wf2[k * C_CLS + 0];
    l1 += z * Wf2[k * C_CLS + 1];
  }
  float mx  = fmaxf(l0, l1);
  float lse = mx + logf(expf(l0 - mx) + expf(l1 - mx));
  out[g * C_CLS + 0] = l0 - lse;
  out[g * C_CLS + 1] = l1 - lse;
}

// ---------------------------------------------------------------------------
static inline int cdiv(int a, int b) { return (a + b - 1) / b; }

extern "C" void kernel_launch(void* const* d_in, const int* in_sizes, int n_in,
                              void* d_out, int out_size, void* d_ws, size_t ws_size,
                              hipStream_t stream) {
  (void)in_sizes; (void)n_in; (void)out_size; (void)ws_size;
  const float* x   = (const float*)d_in[0];
  const int*   ei  = (const int*)d_in[1];
  const int*   bat = (const int*)d_in[2];
  const float* W1  = (const float*)d_in[3];
  const float* as1 = (const float*)d_in[4];
  const float* ad1 = (const float*)d_in[5];
  const float* b1  = (const float*)d_in[6];
  const float* W2  = (const float*)d_in[7];
  const float* as2 = (const float*)d_in[8];
  const float* ad2 = (const float*)d_in[9];
  const float* b2  = (const float*)d_in[10];
  const float* W0  = (const float*)d_in[11];
  const float* b0  = (const float*)d_in[12];
  const float* Wf1 = (const float*)d_in[13];
  const float* bf1 = (const float*)d_in[14];
  const float* Wf2 = (const float*)d_in[15];
  const float* bf2 = (const float*)d_in[16];
  float* out = (float*)d_out;

  char* p = (char*)d_ws;
  auto carve = [&](size_t bytes) {
    char* r = p;
    p += (bytes + 255) & ~(size_t)255;
    return r;
  };
  float*    hA     = (float*)carve((size_t)N_NODES * H_DIM * 4);
  float*    hB     = (float*)carve((size_t)N_NODES * H_DIM * 4);
  unsigned* x16    = (unsigned*)carve((size_t)N_NODES * F_DIM / 2 * 4);
  unsigned* h16    = (unsigned*)carve((size_t)N_NODES * H_DIM / 2 * 4);
  unsigned* w1t    = (unsigned*)carve((size_t)H_DIM * F_DIM / 2 * 4);
  unsigned* w2t    = (unsigned*)carve((size_t)H_DIM * H_DIM / 2 * 4);
  float*    asd    = (float*)carve((size_t)N_NODES * 4);
  float*    add    = (float*)carve((size_t)N_NODES * 4);
  float*    e      = (float*)carve((size_t)(E_EDGES + N_NODES) * 4);
  float*    m      = (float*)carve((size_t)N_NODES * 4);
  float*    denom  = (float*)carve((size_t)N_NODES * 4);
  float*    pooled = (float*)carve((size_t)G_GR * H_DIM * 4);
  float*    cnt    = (float*)carve((size_t)G_GR * 4);
  float*    z1     = (float*)carve((size_t)G_GR * NHID * 4);
  float*    news   = (float*)carve((size_t)G_GR * NHID * 4);
  float*    z2     = (float*)carve((size_t)G_GR * NHID * 4);
  int*      first  = (int*)carve((size_t)G_GR * 4);

  const int TOT = E_EDGES + N_NODES;
  const int TB = 256;
  const int K2a = F_DIM / 2;   // 384 kpairs (GEMM1)
  const int K2b = H_DIM / 2;   // 128 kpairs (GEMM2)
  dim3 blk(TB);
  dim3 gemmGrid(cdiv(N_NODES, 128), NCOL / 128);

  // ---- pre-convert operands to packed bf16 ----
  cvt_pairs<<<cdiv(N_NODES * F_DIM / 2, TB), blk, 0, stream>>>(x, x16, N_NODES * F_DIM / 2);
  cvt_bt<<<cdiv(H_DIM * K2a, TB), blk, 0, stream>>>(W1, w1t, K2a, H_DIM);
  cvt_bt<<<cdiv(H_DIM * K2b, TB), blk, 0, stream>>>(W2, w2t, K2b, H_DIM);

  // ---- GAT layer 1 ----
  gemm_bf16_wmma<<<gemmGrid, blk, 0, stream>>>(x16, w1t, hA, N_NODES, K2a);
  dots_k<<<cdiv(N_NODES * 32, TB), blk, 0, stream>>>(hA, as1, ad1, asd, add);
  initF<<<cdiv(N_NODES, TB), blk, 0, stream>>>(m, N_NODES, -INFINITY);
  initF<<<cdiv(N_NODES, TB), blk, 0, stream>>>(denom, N_NODES, 0.f);
  initF<<<cdiv(N_NODES * H_DIM, TB), blk, 0, stream>>>(hB, N_NODES * H_DIM, 0.f);
  edge_e_max<<<cdiv(TOT, TB), blk, 0, stream>>>(ei, asd, add, e, m);
  edge_exp_sum<<<cdiv(TOT, TB), blk, 0, stream>>>(ei, e, m, denom);
  edge_agg<<<cdiv(TOT * 32, TB), blk, 0, stream>>>(ei, e, hA, hB);
  norm_selu<<<cdiv(N_NODES * H_DIM, TB), blk, 0, stream>>>(hB, denom, b1);

  // ---- GAT layer 2 ----
  cvt_pairs<<<cdiv(N_NODES * H_DIM / 2, TB), blk, 0, stream>>>(hB, h16, N_NODES * H_DIM / 2);
  gemm_bf16_wmma<<<gemmGrid, blk, 0, stream>>>(h16, w2t, hA, N_NODES, K2b);
  dots_k<<<cdiv(N_NODES * 32, TB), blk, 0, stream>>>(hA, as2, ad2, asd, add);
  initF<<<cdiv(N_NODES, TB), blk, 0, stream>>>(m, N_NODES, -INFINITY);
  initF<<<cdiv(N_NODES, TB), blk, 0, stream>>>(denom, N_NODES, 0.f);
  initF<<<cdiv(N_NODES * H_DIM, TB), blk, 0, stream>>>(hB, N_NODES * H_DIM, 0.f);
  edge_e_max<<<cdiv(TOT, TB), blk, 0, stream>>>(ei, asd, add, e, m);
  edge_exp_sum<<<cdiv(TOT, TB), blk, 0, stream>>>(ei, e, m, denom);
  edge_agg<<<cdiv(TOT * 32, TB), blk, 0, stream>>>(ei, e, hA, hB);
  norm_selu<<<cdiv(N_NODES * H_DIM, TB), blk, 0, stream>>>(hB, denom, b2);

  // ---- pooling + head ----
  initF<<<cdiv(G_GR * H_DIM, TB), blk, 0, stream>>>(pooled, G_GR * H_DIM, 0.f);
  initF<<<cdiv(G_GR, TB), blk, 0, stream>>>(cnt, G_GR, 0.f);
  pool_accum<<<cdiv(N_NODES * H_DIM, TB), blk, 0, stream>>>(hB, bat, pooled, cnt);
  pool_finish<<<cdiv(G_GR * H_DIM, TB), blk, 0, stream>>>(pooled, cnt);
  first_idx_k<<<cdiv(N_NODES, TB), blk, 0, stream>>>(bat, first);
  fc1_k<<<cdiv(G_GR * NHID, TB), blk, 0, stream>>>(pooled, Wf1, bf1, z1);
  news_k<<<cdiv(G_GR * NHID, TB), blk, 0, stream>>>(x, first, W0, b0, news);
  fccat_k<<<cdiv(G_GR * NHID, TB), blk, 0, stream>>>(z1, news, Wf1, bf1, z2);
  out_k<<<cdiv(G_GR, TB), blk, 0, stream>>>(z2, Wf2, bf2, out);
}
